// KWS_LSTM_66228395704383
// MI455X (gfx1250) — matrix operand order repaired and checked by reading, not compiled, not run on hardware
//
#include <hip/hip_runtime.h>

// ---------------------------------------------------------------------------
// Quantized LSTM (KWS) for gfx1250.
//  - fp16 WMMA (exact for k/128-quantized weights/activations/states)
//  - X pre-converted to fp16 once (L2-resident; avoids per-step cvt chains)
//  - 2 N-tiles per wave -> two independent WMMA accumulator chains (hides
//    the 5-cycle WMMA->WMMA RAW hazard, halves A traffic per MAC)
//  - fused gate GEMM + LDS tile exchange + pointwise cell update per step
//  - global_prefetch_b8 on the weight stream (CDNA5 prefetch path)
// ---------------------------------------------------------------------------

typedef _Float16 v16h __attribute__((ext_vector_type(16)));
typedef _Float16 v8h  __attribute__((ext_vector_type(8)));
typedef float    v8f  __attribute__((ext_vector_type(8)));
typedef float    v4f  __attribute__((ext_vector_type(4)));

#define T_STEPS 256
#define BATCH   256
#define DIM     128
#define HID     512
#define OUTD    64
#define G4H     2048   // 4*HID

// ---- quantization helpers (bits = 8 everywhere in the reference) ----------
__device__ __forceinline__ float clip8(float x) {
    const float lim = 127.0f / 128.0f;
    return fminf(fmaxf(x, -lim), lim);
}
__device__ __forceinline__ float quant8(float x) {
    // jnp.round is round-half-even -> rintf (v_rndne_f32)
    return rintf(x * 128.0f) * (1.0f / 128.0f);
}
__device__ __forceinline__ float qsig(float x) {
    float s = 1.0f / (1.0f + __expf(-x));
    return quant8(clip8(s));
}
__device__ __forceinline__ float qtanh(float x) {
    return quant8(clip8(tanhf(x)));
}

union AB16 { v16h v; v8h h[2]; };
union F8   { v8f  v; v4f q[2]; };

__device__ __forceinline__ v8f wmma_f16(const AB16& a, const AB16& b, v8f c) {
    return __builtin_amdgcn_wmma_f32_16x16x32_f16(
        /*neg_a=*/false, a.v, /*neg_b=*/false, b.v,
        /*c_mod=*/(short)0, c, /*reuse_a=*/false, /*reuse_b=*/false);
}

// A-operand (16x32 f16): lane<16 -> row=lane, K=[k0+ ..8)+[k0+16..24);
// lane>=16 -> same row, K shifted by 8.  asel = lane>>4.
__device__ __forceinline__ AB16 load_a_h(const _Float16* __restrict__ abase,
                                         int k0, int asel) {
    AB16 a;
    const _Float16* pa = abase + k0 + asel * 8;
    a.h[0] = *(const v8h*)(pa);
    a.h[1] = *(const v8h*)(pa + 16);
    return a;
}
// B-operand (32x16 f16) from [N,K]-major weights: lane's column row-contiguous.
__device__ __forceinline__ AB16 load_b_h(const _Float16* __restrict__ bbase,
                                         int k0, int asel) {
    AB16 b;
    const _Float16* pb = bbase + k0 + asel * 8;
    b.h[0] = *(const v8h*)(pb);
    b.h[1] = *(const v8h*)(pb + 16);
    return b;
}

// ===========================================================================
// Prep kernels (once per launch)
// ===========================================================================
__global__ void k_quant_to_h(const float* __restrict__ in,
                             _Float16* __restrict__ out, int n) {
    int i = blockIdx.x * blockDim.x + threadIdx.x;
    if (i < n) out[i] = (_Float16)quant8(clip8(in[i]));
}

__global__ void k_cvt_to_h(const float* __restrict__ in,
                           _Float16* __restrict__ out, int n) {
    int i = blockIdx.x * blockDim.x + threadIdx.x;
    if (i < n) out[i] = (_Float16)in[i];
}

// vectorized fp32 -> fp16 conversion of the whole input tensor (n % 4 == 0)
__global__ void k_x_to_h(const float* __restrict__ in,
                         _Float16* __restrict__ out, int n4) {
    int i = blockIdx.x * blockDim.x + threadIdx.x;
    if (i < n4) {
        v4f f = *(const v4f*)(in + (size_t)i * 4);
        typedef _Float16 v4h __attribute__((ext_vector_type(4)));
        *(v4h*)(out + (size_t)i * 4) = __builtin_convertvector(f, v4h);
    }
}

__global__ void k_bias_prep(const float* __restrict__ bih,
                            const float* __restrict__ bhh,
                            float* __restrict__ out, int n) {
    int i = blockIdx.x * blockDim.x + threadIdx.x;
    if (i < n) out[i] = quant8(clip8(bih[i])) + quant8(clip8(bhh[i]));
}

__global__ void k_zero_state(_Float16* __restrict__ h0,
                             float* __restrict__ c, int n) {
    int i = blockIdx.x * blockDim.x + threadIdx.x;
    if (i < n) { h0[i] = (_Float16)0.0f; c[i] = 0.0f; }
}

// ===========================================================================
// One LSTM step.
//   grid  = (16 batch tiles, 8 hidden-column groups of 64)
//   block = 256 threads = 8 waves; wave w -> gate (w>>1), col-pair (w&1)
// Each wave computes TWO adjacent 16x16 tiles (independent WMMA chains),
// K = 128 (x@Wih) + 512 (h@Whh) -> 40 WMMAs per wave per step.
// XH16: x pre-converted to fp16 (fast path) vs fp32 convert-in-kernel.
// ===========================================================================
template <bool XH16>
__global__ __launch_bounds__(256) void k_lstm_step(
    const void*     __restrict__ x_t_,   // [B, D] fp16 or fp32
    const _Float16* __restrict__ hprev,  // [B, H] fp16 (quantized, exact)
    const _Float16* __restrict__ Wih,    // [4H, D] fp16
    const _Float16* __restrict__ Whh,    // [4H, H] fp16
    const float*    __restrict__ bias,   // [4H]  (qbih + qbhh)
    float*          __restrict__ cstate, // [B, H] fp32
    _Float16*       __restrict__ hnext)  // [B, H] fp16
{
    __shared__ float gbuf[4][16][65];    // [gate][m][col0..63], padded

    const int lane = threadIdx.x & 31;
    const int wave = threadIdx.x >> 5;
    const int m0   = blockIdx.x * 16;               // batch tile base
    const int hc0  = blockIdx.y * 64;               // hidden-column group base
    const int gate = wave >> 1;
    const int cpair = (wave & 1) * 32;              // 0 or 32 within group
    const int n0   = gate * HID + hc0 + cpair;      // first tile base col in 4H
    const int l15  = lane & 15;
    const int asel = lane >> 4;

    v8f acc0 = {}, acc1 = {};

    const _Float16* bb0 = Whh + (size_t)(n0 + l15) * HID;        // tile 0 B rows
    const _Float16* bb1 = Whh + (size_t)(n0 + 16 + l15) * HID;   // tile 1 B rows

    // -------- input projection: A = x_t, B = Wih --------
    {
        const _Float16* wb0 = Wih + (size_t)(n0 + l15) * DIM;
        const _Float16* wb1 = Wih + (size_t)(n0 + 16 + l15) * DIM;
        if (XH16) {
            const _Float16* abase = (const _Float16*)x_t_ + (size_t)(m0 + l15) * DIM;
            #pragma unroll
            for (int k0 = 0; k0 < DIM; k0 += 32) {
                AB16 a = load_a_h(abase, k0, asel);
                acc0 = wmma_f16(a, load_b_h(wb0, k0, asel), acc0);
                acc1 = wmma_f16(a, load_b_h(wb1, k0, asel), acc1);
            }
        } else {
            const float* abase = (const float*)x_t_ + (size_t)(m0 + l15) * DIM;
            #pragma unroll
            for (int k0 = 0; k0 < DIM; k0 += 32) {
                AB16 a;
                const float* pa = abase + k0 + asel * 8;
                F8 f0, f1;
                f0.q[0] = *(const v4f*)(pa);
                f0.q[1] = *(const v4f*)(pa + 4);
                f1.q[0] = *(const v4f*)(pa + 16);
                f1.q[1] = *(const v4f*)(pa + 20);
                a.h[0] = __builtin_convertvector(f0.v, v8h);
                a.h[1] = __builtin_convertvector(f1.v, v8h);
                acc0 = wmma_f16(a, load_b_h(wb0, k0, asel), acc0);
                acc1 = wmma_f16(a, load_b_h(wb1, k0, asel), acc1);
            }
        }
    }

    // -------- recurrent projection: A = hprev (fp16), B = Whh --------
    {
        const _Float16* abase = hprev + (size_t)(m0 + l15) * HID;
        #pragma unroll 4
        for (int k0 = 0; k0 < HID; k0 += 32) {
            if (k0 + 64 < HID) {   // pull next weight chunk toward WGP (CDNA5)
                __builtin_prefetch((const void*)(bb0 + k0 + 64), 0, 1);
                __builtin_prefetch((const void*)(bb1 + k0 + 64), 0, 1);
            }
            AB16 a = load_a_h(abase, k0, asel);
            acc0 = wmma_f16(a, load_b_h(bb0, k0, asel), acc0);
            acc1 = wmma_f16(a, load_b_h(bb1, k0, asel), acc1);
        }
    }

    // -------- scatter gate tiles to LDS (C/D: row r+8*asel, col lane&15) ----
    {
        #pragma unroll
        for (int r = 0; r < 8; ++r) {
            const int m = r + asel * 8;
            gbuf[gate][m][cpair + l15]      = acc0[r];
            gbuf[gate][m][cpair + 16 + l15] = acc1[r];
        }
    }
    __syncthreads();

    // -------- pointwise LSTM cell update (16x64 elems, 4 per thread) --------
    for (int e = threadIdx.x; e < 16 * 64; e += 256) {
        const int m    = e >> 6;
        const int col  = e & 63;
        const int hcol = hc0 + col;
        float iv = qsig (gbuf[0][m][col] + bias[0 * HID + hcol]);
        float fv = qsig (gbuf[1][m][col] + bias[1 * HID + hcol]);
        float gv = qtanh(gbuf[2][m][col] + bias[2 * HID + hcol]);
        float ov = qsig (gbuf[3][m][col] + bias[3 * HID + hcol]);
        const size_t gidx = (size_t)(m0 + m) * HID + hcol;
        float cv = quant8(clip8(fv * cstate[gidx] + iv * gv));
        cstate[gidx] = cv;
        float hv = quant8(clip8(ov * tanhf(cv)));
        hnext[gidx] = (_Float16)hv;
    }
}

// ===========================================================================
// Final projection: out = quant8(sigmoid(h_last @ Woutᵀ + b_out))
// 4 blocks x 8 waves x 2 tiles = 64 tiles (16 m x 4 n), K=512.
// ===========================================================================
__global__ __launch_bounds__(256) void k_out_proj(
    const _Float16* __restrict__ hlast,  // [B, H] fp16
    const _Float16* __restrict__ WoutH,  // [O, H] fp16
    const float*    __restrict__ bout,   // [O]
    float*          __restrict__ out)    // [B, O]
{
    const int lane = threadIdx.x & 31;
    const int wave = threadIdx.x >> 5;
    const int pair = blockIdx.x * 8 + wave;   // 0..31
    const int m0 = (pair >> 1) * 16;
    const int n0 = (pair & 1) * 32;
    const int l15  = lane & 15;
    const int asel = lane >> 4;

    v8f acc0 = {}, acc1 = {};
    const _Float16* abase = hlast + (size_t)(m0 + l15) * HID;
    const _Float16* bb0 = WoutH + (size_t)(n0 + l15) * HID;
    const _Float16* bb1 = WoutH + (size_t)(n0 + 16 + l15) * HID;
    #pragma unroll 4
    for (int k0 = 0; k0 < HID; k0 += 32) {
        AB16 a = load_a_h(abase, k0, asel);
        acc0 = wmma_f16(a, load_b_h(bb0, k0, asel), acc0);
        acc1 = wmma_f16(a, load_b_h(bb1, k0, asel), acc1);
    }

    const float bv0 = bout[n0 + l15];
    const float bv1 = bout[n0 + 16 + l15];
    #pragma unroll
    for (int r = 0; r < 8; ++r) {
        const int m = m0 + r + asel * 8;
        out[(size_t)m * OUTD + n0 + l15]      = qsig(acc0[r] + bv0);
        out[(size_t)m * OUTD + n0 + 16 + l15] = qsig(acc1[r] + bv1);
    }
}

// ===========================================================================
// Host-side launch
// ===========================================================================
extern "C" void kernel_launch(void* const* d_in, const int* in_sizes, int n_in,
                              void* d_out, int out_size, void* d_ws, size_t ws_size,
                              hipStream_t stream)
{
    const float* inputs = (const float*)d_in[0];  // [T,B,D]
    const float* w_ih   = (const float*)d_in[1];  // [4H,D]
    const float* w_hh   = (const float*)d_in[2];  // [4H,H]
    const float* b_ih   = (const float*)d_in[3];  // [4H]
    const float* b_hh   = (const float*)d_in[4];  // [4H]
    const float* w_out  = (const float*)d_in[5];  // [O,H]
    const float* b_out  = (const float*)d_in[6];  // [O]

    char* ws = (char*)d_ws;
    size_t off = 0;
    _Float16* WihH  = (_Float16*)(ws + off); off += (size_t)G4H * DIM * 2;   // 512 KB
    _Float16* WhhH  = (_Float16*)(ws + off); off += (size_t)G4H * HID * 2;   // 2 MB
    _Float16* WoutH = (_Float16*)(ws + off); off += (size_t)OUTD * HID * 2;  // 64 KB
    float*    biasW = (float*)   (ws + off); off += (size_t)G4H * 4;         // 8 KB
    _Float16* hbuf0 = (_Float16*)(ws + off); off += (size_t)BATCH * HID * 2; // 256 KB
    _Float16* hbuf1 = (_Float16*)(ws + off); off += (size_t)BATCH * HID * 2; // 256 KB
    float*    cbuf  = (float*)   (ws + off); off += (size_t)BATCH * HID * 4; // 512 KB
    _Float16* Xh    = (_Float16*)(ws + off);
    const size_t xh_bytes = (size_t)T_STEPS * BATCH * DIM * 2;               // 16 MB
    const bool use_xh = (ws_size >= off + xh_bytes);

    // ---- prep: quantize weights to exact fp16, fold biases, zero state ----
    k_quant_to_h<<<(G4H * DIM + 255) / 256, 256, 0, stream>>>(w_ih, WihH, G4H * DIM);
    k_quant_to_h<<<(G4H * HID + 255) / 256, 256, 0, stream>>>(w_hh, WhhH, G4H * HID);
    k_cvt_to_h  <<<(OUTD * HID + 255) / 256, 256, 0, stream>>>(w_out, WoutH, OUTD * HID);
    k_bias_prep <<<(G4H + 255) / 256, 256, 0, stream>>>(b_ih, b_hh, biasW, G4H);
    k_zero_state<<<(BATCH * HID + 255) / 256, 256, 0, stream>>>(hbuf0, cbuf, BATCH * HID);
    if (use_xh) {
        const int n4 = T_STEPS * BATCH * DIM / 4;
        k_x_to_h<<<(n4 + 255) / 256, 256, 0, stream>>>(inputs, Xh, n4);
    }

    // ---- sequential recurrence: 256 fused step kernels (ping-pong h) ------
    _Float16* hb[2] = { hbuf0, hbuf1 };
    for (int t = 0; t < T_STEPS; ++t) {
        const dim3 grid(16, 8);
        if (use_xh) {
            k_lstm_step<true><<<grid, 256, 0, stream>>>(
                (const void*)(Xh + (size_t)t * BATCH * DIM),
                hb[t & 1], WihH, WhhH, biasW, cbuf, hb[(t + 1) & 1]);
        } else {
            k_lstm_step<false><<<grid, 256, 0, stream>>>(
                (const void*)(inputs + (size_t)t * BATCH * DIM),
                hb[t & 1], WihH, WhhH, biasW, cbuf, hb[(t + 1) & 1]);
        }
    }

    // ---- final projection (h_last ends up in hbuf0 since T is even) -------
    k_out_proj<<<4, 256, 0, stream>>>(hb[T_STEPS & 1], WoutH, b_out, (float*)d_out);
}
